// SAttn_78546361909990
// MI455X (gfx1250) — matrix operand order repaired and checked by reading, not compiled, not run on hardware
//
#include <hip/hip_runtime.h>
#include <hip/hip_bf16.h>
#include <math.h>

#define DIM   1024
#define HEADS 8
#define SPLIT 128
#define BATCH 4
#define SEQ   2048
#define ROWS  (BATCH*SEQ)   // 8192
#define BHTOT (BATCH*HEADS) // 32

typedef __attribute__((ext_vector_type(16))) __bf16 v16bf;
typedef __attribute__((ext_vector_type(8)))  __bf16 v8bf;
typedef __attribute__((ext_vector_type(8)))  float  v8f;

__device__ __forceinline__ unsigned short f2bfbits(float f){
    unsigned u = __builtin_bit_cast(unsigned, f);
    unsigned r = u + 0x7FFFu + ((u >> 16) & 1u);
    return (unsigned short)(r >> 16);
}
__device__ __forceinline__ __bf16 f2bf(float f){
    unsigned short h = f2bfbits(f);
    return __builtin_bit_cast(__bf16, h);
}

__device__ __forceinline__ v8f wmma_bf16(v16bf a, v16bf b, v8f c){
    return __builtin_amdgcn_wmma_f32_16x16x32_bf16(false, a, false, b, (short)0, c, false, false);
}
__device__ __forceinline__ v8f vzero8(){
    v8f c;
    #pragma unroll
    for (int i = 0; i < 8; ++i) c[i] = 0.f;
    return c;
}

// ---- fragment loaders (A and B both use contiguous k = 16*half + s) ----
// A 16x32 bf16: lanes 0-15 = rows; K elems {half*16+0..15}
__device__ __forceinline__ v16bf load_a_bf16(const __bf16* row, int k0, int half){
    v8bf lo = *(const v8bf*)(row + k0 + half*16);
    v8bf hi = *(const v8bf*)(row + k0 + half*16 + 8);
    v16bf a;
    #pragma unroll
    for (int i = 0; i < 8; ++i){ a[i] = lo[i]; a[i+8] = hi[i]; }
    return a;
}
// B 32x16 bf16 from Bt[N][K]: lane = col; lanes16-31 take K 16..31
__device__ __forceinline__ v16bf load_b_bf16(const __bf16* btrow, int k0, int half){
    return *(const v16bf*)(btrow + k0 + half*16);
}

// 32x32 output tile per wave: 2x2 accumulators, 4 frag loads -> 4 WMMAs per k-step
__device__ __forceinline__ void gemm_32x32(const __bf16* arow, const __bf16* brow,
                                           int lda, int ldb, int Kdim, int half,
                                           v8f c[2][2]){
    const __bf16* arow1 = arow + (size_t)16 * lda;
    const __bf16* brow1 = brow + (size_t)16 * ldb;
    for (int k0 = 0; k0 < Kdim; k0 += 32){
        v16bf a0 = load_a_bf16(arow,  k0, half);
        v16bf a1 = load_a_bf16(arow1, k0, half);
        v16bf b0 = load_b_bf16(brow,  k0, half);
        v16bf b1 = load_b_bf16(brow1, k0, half);
        c[0][0] = wmma_bf16(a0, b0, c[0][0]);
        c[0][1] = wmma_bf16(a0, b1, c[0][1]);
        c[1][0] = wmma_bf16(a1, b0, c[1][0]);
        c[1][1] = wmma_bf16(a1, b1, c[1][1]);
    }
}

// ---- flat f32 -> bf16 convert (vectorized: 4 elems/thread) ----
__global__ void __launch_bounds__(256)
k_cvt(const float* __restrict__ src, unsigned short* __restrict__ dst, int n4){
    int idx = blockIdx.x * 256 + threadIdx.x;
    if (idx >= n4) return;
    float4 v = *(const float4*)(src + (size_t)idx * 4);
    ushort4 o;
    o.x = f2bfbits(v.x); o.y = f2bfbits(v.y); o.z = f2bfbits(v.z); o.w = f2bfbits(v.w);
    *(ushort4*)(dst + (size_t)idx * 4) = o;
}

// ---- weight convert + transpose:  dst[n*K + k] = bf16(src[k*N + n]) ----
__global__ void k_cvtT(const float* __restrict__ src, unsigned short* __restrict__ dst,
                       int K, int N){
    int idx = blockIdx.x * 256 + threadIdx.x;
    if (idx >= K * N) return;
    int n = idx / K, k = idx - n * K;
    dst[idx] = f2bfbits(src[(size_t)k * N + n]);
}
__global__ void k_cvtT_batch(const float* __restrict__ src, unsigned short* __restrict__ dst){
    int idx = blockIdx.x * 256 + threadIdx.x;   // 0..16383
    int n = idx >> 7, k = idx & 127;
    size_t off = (size_t)blockIdx.y * (SPLIT * SPLIT);
    dst[off + idx] = f2bfbits(src[off + (size_t)k * SPLIT + n]);
}

// ---- GEMM1: mod = x @ W_in  (A bf16 pre-converted, Bt bf16) ----
__global__ void __launch_bounds__(128)
k_gemm_in(const unsigned short* __restrict__ Xb, const unsigned short* __restrict__ WinT,
          float* __restrict__ Mod){
    const int lane = threadIdx.x, half = lane >> 4, ln = lane & 15;
    const int m0 = blockIdx.x * 32;
    const int n0 = (blockIdx.y * 4 + threadIdx.y) * 32;
    const __bf16* arow = (const __bf16*)Xb   + (size_t)(m0 + ln) * DIM;
    const __bf16* brow = (const __bf16*)WinT + (size_t)(n0 + ln) * DIM;
    v8f c[2][2] = {{vzero8(), vzero8()}, {vzero8(), vzero8()}};
    gemm_32x32(arow, brow, DIM, DIM, DIM, half, c);
    #pragma unroll
    for (int mi = 0; mi < 2; ++mi)
    #pragma unroll
    for (int ni = 0; ni < 2; ++ni)
    #pragma unroll
    for (int r = 0; r < 8; ++r)
        Mod[(size_t)(m0 + mi*16 + r + 8*half) * DIM + n0 + ni*16 + ln] = c[mi][ni][r];
}

// ---- LayerNorm row kernel: dst(bf16) = LN(src) * gamma + beta ----
__global__ void __launch_bounds__(256)
k_ln(const float* __restrict__ src, unsigned short* __restrict__ dst,
     const float* __restrict__ gamma, const float* __restrict__ beta){
    __shared__ float red[256];
    const int row = blockIdx.x, t = threadIdx.x;
    const float* x = src + (size_t)row * DIM;
    float4 v = *(const float4*)(x + t * 4);
    float s  = v.x + v.y + v.z + v.w;
    float s2 = v.x*v.x + v.y*v.y + v.z*v.z + v.w*v.w;
    red[t] = s; __syncthreads();
    for (int o = 128; o > 0; o >>= 1){ if (t < o) red[t] += red[t + o]; __syncthreads(); }
    float mean = red[0] * (1.f / DIM); __syncthreads();
    red[t] = s2; __syncthreads();
    for (int o = 128; o > 0; o >>= 1){ if (t < o) red[t] += red[t + o]; __syncthreads(); }
    float var  = red[0] * (1.f / DIM) - mean * mean;
    float rstd = rsqrtf(var + 1e-6f);
    unsigned short* d = dst + (size_t)row * DIM;
    const float xv[4] = {v.x, v.y, v.z, v.w};
    #pragma unroll
    for (int i = 0; i < 4; ++i){
        int col = t * 4 + i;
        d[col] = f2bfbits((xv[i] - mean) * rstd * gamma[col] + beta[col]);
    }
}

// ---- QKV: per-head projections; Q,K row-major [bh,s,e], V transposed [bh,e,s] ----
__global__ void __launch_bounds__(128)
k_qkv(const unsigned short* __restrict__ H,
      const unsigned short* __restrict__ WqT, const unsigned short* __restrict__ WkT,
      const unsigned short* __restrict__ WvT,
      unsigned short* __restrict__ Q, unsigned short* __restrict__ K,
      unsigned short* __restrict__ VT){
    const int lane = threadIdx.x, half = lane >> 4, ln = lane & 15;
    const int m0 = blockIdx.x * 32;
    const int n0 = threadIdx.y * 32;            // 4 waves cover N=128
    const int z = blockIdx.z, proj = z >> 3, head = z & 7;
    const unsigned short* W = (proj == 0) ? WqT : (proj == 1 ? WkT : WvT);
    const __bf16* arow = (const __bf16*)H + (size_t)(m0 + ln) * DIM + head * SPLIT;
    const __bf16* brow = (const __bf16*)W + (size_t)head * SPLIT * SPLIT + (size_t)(n0 + ln) * SPLIT;
    v8f c[2][2] = {{vzero8(), vzero8()}, {vzero8(), vzero8()}};
    gemm_32x32(arow, brow, DIM, SPLIT, SPLIT, half, c);
    #pragma unroll
    for (int mi = 0; mi < 2; ++mi)
    #pragma unroll
    for (int ni = 0; ni < 2; ++ni)
    #pragma unroll
    for (int r = 0; r < 8; ++r){
        int m = m0 + mi*16 + r + 8*half;
        int bb = m >> 11, s = m & 2047;
        int bh = bb * HEADS + head;
        int col = n0 + ni*16 + ln;
        unsigned short bits = f2bfbits(c[mi][ni][r]);
        if (proj == 2) VT[((size_t)bh * SPLIT + col) * SEQ + s] = bits;
        else {
            unsigned short* O = (proj == 0) ? Q : K;
            O[((size_t)bh * SEQ + s) * SPLIT + col] = bits;
        }
    }
}

// ---- Flash attention: wave handles 16 queries x all keys; Mod += softmax(QK^T)V ----
__global__ void __launch_bounds__(128)
k_attn(const unsigned short* __restrict__ Qb, const unsigned short* __restrict__ Kb,
       const unsigned short* __restrict__ VTb, float* __restrict__ Mod){
    __shared__ __align__(32) __bf16 Pbuf[4][16 * 32];
    const int lane = threadIdx.x, half = lane >> 4, ln = lane & 15, w = threadIdx.y;
    const int qt = blockIdx.x * 4 + w;
    const int bh = blockIdx.y;
    const int m0 = qt * 16;
    __bf16* P = Pbuf[w];

    const __bf16* qrow = (const __bf16*)Qb + ((size_t)bh * SEQ + m0 + ln) * SPLIT;
    v16bf aq[4];
    #pragma unroll
    for (int j = 0; j < 4; ++j) aq[j] = load_a_bf16(qrow, j * 32, half);

    float mrow[8], lrow[8];
    v8f o[8];
    #pragma unroll
    for (int r = 0; r < 8; ++r){ mrow[r] = -1e30f; lrow[r] = 0.f; }
    #pragma unroll
    for (int f = 0; f < 8; ++f) o[f] = vzero8();

    for (int kc = 0; kc < SEQ; kc += 32){
        v8f s0 = vzero8(), s1 = vzero8();
        {
            const __bf16* krow0 = (const __bf16*)Kb + ((size_t)bh * SEQ + kc + ln) * SPLIT;
            const __bf16* krow1 = krow0 + (size_t)16 * SPLIT;
            #pragma unroll
            for (int j = 0; j < 4; ++j){
                v16bf b0 = load_b_bf16(krow0, j * 32, half);
                s0 = wmma_bf16(aq[j], b0, s0);
                v16bf b1 = load_b_bf16(krow1, j * 32, half);
                s1 = wmma_bf16(aq[j], b1, s1);
            }
        }
        // online softmax, row reductions across the 16 lanes of each half
        #pragma unroll
        for (int r = 0; r < 8; ++r){
            float mx = fmaxf(s0[r], s1[r]);
            #pragma unroll
            for (int off = 8; off >= 1; off >>= 1) mx = fmaxf(mx, __shfl_xor(mx, off, 32));
            float mnew  = fmaxf(mrow[r], mx);
            float scale = __expf(mrow[r] - mnew);
            float p0 = __expf(s0[r] - mnew);
            float p1 = __expf(s1[r] - mnew);
            float ls = p0 + p1;
            #pragma unroll
            for (int off = 8; off >= 1; off >>= 1) ls += __shfl_xor(ls, off, 32);
            lrow[r] = lrow[r] * scale + ls;
            mrow[r] = mnew;
            #pragma unroll
            for (int f = 0; f < 8; ++f) o[f][r] *= scale;
            P[(r + 8*half) * 32 + ln]      = f2bf(p0);
            P[(r + 8*half) * 32 + 16 + ln] = f2bf(p1);
        }
        asm volatile("s_wait_dscnt 0" ::: "memory");
        v16bf ap = load_a_bf16(P + ln * 32, 0, half);
        #pragma unroll
        for (int f = 0; f < 8; ++f){
            const __bf16* vrow = (const __bf16*)VTb + ((size_t)bh * SPLIT + f*16 + ln) * SEQ;
            v16bf b = *(const v16bf*)(vrow + kc + half * 16);
            o[f] = wmma_bf16(ap, b, o[f]);
        }
    }
    // epilogue: normalize and residual-add into Mod (disjoint per (bh, qtile))
    const int bb = bh >> 3, head = bh & 7;
    #pragma unroll
    for (int r = 0; r < 8; ++r){
        float inv = 1.f / lrow[r];
        int s = m0 + r + 8*half;
        size_t base = ((size_t)(bb * SEQ + s)) * DIM + head * SPLIT;
        #pragma unroll
        for (int f = 0; f < 8; ++f){
            size_t idx = base + f*16 + ln;
            Mod[idx] += o[f][r] * inv;
        }
    }
}

// ---- FFN GEMM with fused bias + ELU + residual:  out = mod + elu(t@Wf + bf) ----
__global__ void __launch_bounds__(128)
k_gemm_ffn(const unsigned short* __restrict__ T, const unsigned short* __restrict__ WfT,
           const float* __restrict__ bias, const float* __restrict__ Mod,
           float* __restrict__ Out){
    const int lane = threadIdx.x, half = lane >> 4, ln = lane & 15;
    const int m0 = blockIdx.x * 32;
    const int n0 = (blockIdx.y * 4 + threadIdx.y) * 32;
    const __bf16* arow = (const __bf16*)T   + (size_t)(m0 + ln) * DIM;
    const __bf16* brow = (const __bf16*)WfT + (size_t)(n0 + ln) * DIM;
    v8f c[2][2] = {{vzero8(), vzero8()}, {vzero8(), vzero8()}};
    gemm_32x32(arow, brow, DIM, DIM, DIM, half, c);
    #pragma unroll
    for (int mi = 0; mi < 2; ++mi)
    #pragma unroll
    for (int ni = 0; ni < 2; ++ni)
    #pragma unroll
    for (int r = 0; r < 8; ++r){
        int m = m0 + mi*16 + r + 8*half, col = n0 + ni*16 + ln;
        float y = c[mi][ni][r] + bias[col];
        float e = (y > 0.f) ? y : expm1f(y);
        size_t idx = (size_t)m * DIM + col;
        Out[idx] = Mod[idx] + e;
    }
}

extern "C" void kernel_launch(void* const* d_in, const int* in_sizes, int n_in,
                              void* d_out, int out_size, void* d_ws, size_t ws_size,
                              hipStream_t stream){
    (void)in_sizes; (void)n_in; (void)out_size; (void)ws_size;
    const float* x     = (const float*)d_in[0];
    const float* W_in  = (const float*)d_in[1];
    const float* gamma = (const float*)d_in[2];
    const float* beta  = (const float*)d_in[3];
    const float* Wq    = (const float*)d_in[4];
    const float* Wk    = (const float*)d_in[5];
    const float* Wv    = (const float*)d_in[6];
    const float* Wf    = (const float*)d_in[7];
    const float* bf_   = (const float*)d_in[8];
    float* out = (float*)d_out;

    char* ws = (char*)d_ws;
    float* mod            = (float*)ws;          ws += (size_t)ROWS * DIM * 4;
    unsigned short* hbuf  = (unsigned short*)ws; ws += (size_t)ROWS * DIM * 2;   // also holds bf16(x)
    unsigned short* winT  = (unsigned short*)ws; ws += (size_t)DIM * DIM * 2;
    unsigned short* wfT   = (unsigned short*)ws; ws += (size_t)DIM * DIM * 2;
    unsigned short* wqT   = (unsigned short*)ws; ws += (size_t)HEADS * SPLIT * SPLIT * 2;
    unsigned short* wkT   = (unsigned short*)ws; ws += (size_t)HEADS * SPLIT * SPLIT * 2;
    unsigned short* wvT   = (unsigned short*)ws; ws += (size_t)HEADS * SPLIT * SPLIT * 2;
    unsigned short* qb    = (unsigned short*)ws; ws += (size_t)BHTOT * SEQ * SPLIT * 2;
    unsigned short* kb    = (unsigned short*)ws; ws += (size_t)BHTOT * SEQ * SPLIT * 2;
    unsigned short* vtb   = (unsigned short*)ws; ws += (size_t)BHTOT * SEQ * SPLIT * 2;

    dim3 blk(32, 4);
    // weights -> bf16 (transposed), x -> bf16 (into hbuf; dead until LN1)
    k_cvtT<<<dim3((DIM * DIM + 255) / 256), 256, 0, stream>>>(W_in, winT, DIM, DIM);
    k_cvtT<<<dim3((DIM * DIM + 255) / 256), 256, 0, stream>>>(Wf, wfT, DIM, DIM);
    k_cvtT_batch<<<dim3(64, HEADS), 256, 0, stream>>>(Wq, wqT);
    k_cvtT_batch<<<dim3(64, HEADS), 256, 0, stream>>>(Wk, wkT);
    k_cvtT_batch<<<dim3(64, HEADS), 256, 0, stream>>>(Wv, wvT);
    k_cvt<<<dim3((ROWS * DIM / 4 + 255) / 256), 256, 0, stream>>>(x, hbuf, ROWS * DIM / 4);

    k_gemm_in<<<dim3(ROWS / 32, DIM / 128), blk, 0, stream>>>(hbuf, winT, mod);
    k_ln<<<ROWS, 256, 0, stream>>>(mod, hbuf, gamma, beta);
    k_qkv<<<dim3(ROWS / 32, 1, 24), blk, 0, stream>>>(hbuf, wqT, wkT, wvT, qb, kb, vtb);
    k_attn<<<dim3(SEQ / 64, BHTOT), blk, 0, stream>>>(qb, kb, vtb, mod);
    k_ln<<<ROWS, 256, 0, stream>>>(mod, hbuf, gamma, beta);
    k_gemm_ffn<<<dim3(ROWS / 32, DIM / 128), blk, 0, stream>>>(hbuf, wfT, bf_, mod, out);
}
